// TrafficGATCritic_81724637708834
// MI455X (gfx1250) — matrix operand order repaired and checked by reading, not compiled, not run on hardware
//
#include <hip/hip_runtime.h>
#include <hip/hip_bf16.h>
#include <math.h>

// ---------------- constants for this problem ----------------
#define F_IN   70
#define KP1    96          // F_IN padded to multiple of 32 for WMMA K
#define HID    256
#define HEADS  4
#define HC1    (HEADS*HID) // 1024
#define NEG_SLOPE 0.2f

typedef __bf16 v16bf __attribute__((ext_vector_type(16)));
typedef __bf16 v8bf  __attribute__((ext_vector_type(8)));
typedef float  v8f   __attribute__((ext_vector_type(8)));

// ---------------- small utility kernels ----------------
__global__ void fill_f32(float* p, float v, int n) {
    int i = blockIdx.x * blockDim.x + threadIdx.x;
    if (i < n) p[i] = v;
}
__global__ void fill_u32(unsigned* p, unsigned v, int n) {
    int i = blockIdx.x * blockDim.x + threadIdx.x;
    if (i < n) p[i] = v;
}

// node_features [N,F_IN] f32 -> [N,KP1] bf16 (zero padded K)
__global__ void cast_pad_x(const float* __restrict__ x, __hip_bfloat16* __restrict__ xb, int N) {
    int i = blockIdx.x * blockDim.x + threadIdx.x;
    if (i >= N * KP1) return;
    int n = i / KP1, k = i - n * KP1;
    float v = (k < F_IN) ? x[(size_t)n * F_IN + k] : 0.0f;
    xb[i] = __float2bfloat16(v);
}

// W [K0,NC] f32 (row-major) -> Wt [NC,Kpad] bf16 (row-major == transposed, padded)
__global__ void transpose_cast_w(const float* __restrict__ w, __hip_bfloat16* __restrict__ wt,
                                 int K0, int Kpad, int NC) {
    int i = blockIdx.x * blockDim.x + threadIdx.x;
    if (i >= NC * Kpad) return;
    int o = i / Kpad, k = i - o * Kpad;
    float v = (k < K0) ? w[(size_t)k * NC + o] : 0.0f;
    wt[i] = __float2bfloat16(v);
}

// ---------------- WMMA bf16 GEMM: C[M,Nc] = A[M,K] * Bt[N,K]^T ----------------
// One wave computes a 16x64 output strip (4 accumulators reuse the A fragment).
// A fragment (16x32 bf16, ISA 7.12.2): lanes 0-15 row=lane, K={0..7,16..23};
//   lanes 16-31 row=lane-16, K={8..15,24..31}.
// B fragment (32x16 bf16): lanes 0-15 col=lane K={0..7,16..23}; lanes 16-31 col K={8..15,24..31}.
// C/D (16x16 f32): vgpr r, lane l -> row = r + 8*(l>=16), col = l&15.
__global__ void gemm_wmma_bf16(const __bf16* __restrict__ A, const __bf16* __restrict__ Bt,
                               float* __restrict__ C, int M, int Nc, int K) {
    const int lane = threadIdx.x & 31;
    const int wave = (blockIdx.x * blockDim.x + threadIdx.x) >> 5;
    const int tilesN4 = Nc >> 6;                 // groups of 64 output columns
    const int tilesM  = M >> 4;
    if (wave >= tilesM * tilesN4) return;        // wave-uniform: EXEC stays all-ones
    const int tm  = wave / tilesN4;
    const int tn4 = wave - tm * tilesN4;
    const int hi  = lane >> 4;                   // lane-half select
    const int row = tm * 16 + (lane & 15);
    const int colbase = tn4 * 64 + (lane & 15);

    v8f acc0 = {}, acc1 = {}, acc2 = {}, acc3 = {};
    for (int k0 = 0; k0 < K; k0 += 32) {
        const __bf16* ap = A + (size_t)row * K + k0;
        v8bf alo = *(const v8bf*)(ap + hi * 8);
        v8bf ahi = *(const v8bf*)(ap + 16 + hi * 8);
        v16bf a = __builtin_shufflevector(alo, ahi, 0,1,2,3,4,5,6,7,8,9,10,11,12,13,14,15);

        #pragma unroll
        for (int s = 0; s < 4; ++s) {
            const __bf16* bp = Bt + (size_t)(colbase + s * 16) * K + k0 + hi * 8;
            v8bf blo = *(const v8bf*)(bp);
            v8bf bhi = *(const v8bf*)(bp + 16);
            v16bf b = __builtin_shufflevector(blo, bhi, 0,1,2,3,4,5,6,7,8,9,10,11,12,13,14,15);
            v8f* acc = (s == 0) ? &acc0 : (s == 1) ? &acc1 : (s == 2) ? &acc2 : &acc3;
            *acc = __builtin_amdgcn_wmma_f32_16x16x32_bf16(
                false, a, false, b, (short)0, *acc, false, false);
        }
    }
    #pragma unroll
    for (int s = 0; s < 4; ++s) {
        const v8f acc = (s == 0) ? acc0 : (s == 1) ? acc1 : (s == 2) ? acc2 : acc3;
        #pragma unroll
        for (int r = 0; r < 8; ++r)
            C[(size_t)(tm * 16 + r + hi * 8) * Nc + colbase + s * 16] = acc[r];
    }
}

// ---------------- attention dot products: a_s/a_d [N,H] ----------------
// block = 32*H threads; wave w handles head w of node blockIdx.x
__global__ void att_dots(const float* __restrict__ h, const float* __restrict__ asrc,
                         const float* __restrict__ adst, int H, int C,
                         float* __restrict__ out_s, float* __restrict__ out_d) {
    int n = blockIdx.x;
    int w = threadIdx.x >> 5;
    int lane = threadIdx.x & 31;
    const float* hp = h + (size_t)n * H * C + (size_t)w * C;
    float ps = 0.f, pd = 0.f;
    for (int c = lane; c < C; c += 32) {
        float v = hp[c];
        ps += v * asrc[w * C + c];
        pd += v * adst[w * C + c];
    }
    #pragma unroll
    for (int m = 16; m; m >>= 1) { ps += __shfl_xor(ps, m, 32); pd += __shfl_xor(pd, m, 32); }
    if (lane == 0) { out_s[n * H + w] = ps; out_d[n * H + w] = pd; }
}

// ---------------- segment softmax over destination nodes ----------------
__device__ __forceinline__ unsigned fenc(float f) {          // order-preserving float->uint
    int i = __float_as_int(f);
    return (unsigned)(i ^ ((i >> 31) | 0x80000000));
}
__device__ __forceinline__ float fdec(unsigned u) {
    int i = (u & 0x80000000u) ? (int)(u ^ 0x80000000u) : ~(int)u;
    return __int_as_float(i);
}
__device__ __forceinline__ void edge_sd(const int* ei, int E, int j, int& s, int& d) {
    if (j < E) { s = ei[j]; d = ei[E + j]; } else { s = d = j - E; }
}

__global__ void edge_logit_max(const float* __restrict__ as, const float* __restrict__ ad,
                               const int* __restrict__ ei, int E, int N, int H,
                               float* __restrict__ ebuf, unsigned* __restrict__ menc) {
    int tid = blockIdx.x * blockDim.x + threadIdx.x;
    if (tid >= (E + N) * H) return;
    int j = tid / H, h = tid - j * H, s, d;
    edge_sd(ei, E, j, s, d);
    float e = as[s * H + h] + ad[d * H + h];
    e = (e > 0.f) ? e : NEG_SLOPE * e;                       // leaky_relu
    ebuf[tid] = e;
    atomicMax(&menc[d * H + h], fenc(e));
}

__global__ void edge_exp_sum(const int* __restrict__ ei, int E, int N, int H,
                             float* __restrict__ ebuf, const unsigned* __restrict__ menc,
                             float* __restrict__ den) {
    int tid = blockIdx.x * blockDim.x + threadIdx.x;
    if (tid >= (E + N) * H) return;
    int j = tid / H, h = tid - j * H, s, d;
    edge_sd(ei, E, j, s, d);
    float m = fdec(menc[d * H + h]);
    float ex = expf(ebuf[tid] - m);
    ebuf[tid] = ex;
    atomicAdd(&den[d * H + h], ex);
}

// out[dst] += h[src] * alpha ; one block per edge, float4 per thread
__global__ void aggregate(const float* __restrict__ hsrc, const float* __restrict__ ebuf,
                          const float* __restrict__ den, const int* __restrict__ ei,
                          int E, int H, int C, float* __restrict__ out) {
    int j = blockIdx.x, s, d;
    edge_sd(ei, E, j, s, d);
    int HC = H * C;
    int c4 = threadIdx.x * 4;
    int h  = c4 / C;
    float alpha = ebuf[(size_t)j * H + h] / (den[d * H + h] + 1e-16f);
    const float4 v = *(const float4*)(hsrc + (size_t)s * HC + c4);
    float* o = out + (size_t)d * HC + c4;
    atomicAdd(o + 0, v.x * alpha);
    atomicAdd(o + 1, v.y * alpha);
    atomicAdd(o + 2, v.z * alpha);
    atomicAdd(o + 3, v.w * alpha);
}

// x2b = bf16(relu(out1 + b1)) elementwise, [N,HC]
__global__ void bias_relu_cast(const float* __restrict__ src, const float* __restrict__ b,
                               __hip_bfloat16* __restrict__ dst, int total, int HC) {
    int i = blockIdx.x * blockDim.x + threadIdx.x;
    if (i >= total) return;
    float v = src[i] + b[i % HC];
    dst[i] = __float2bfloat16(fmaxf(v, 0.f));
}

// pooled[c] += mean_n relu(out2[n,c] + b2[c])
__global__ void pool_mean_relu(const float* __restrict__ out2, const float* __restrict__ b2,
                               int N, float* __restrict__ pooled) {
    int t = threadIdx.x;                                     // 256 channels
    float acc = 0.f;
    float bb = b2[t];
    for (int n = blockIdx.x; n < N; n += gridDim.x)
        acc += fmaxf(out2[(size_t)n * HID + t] + bb, 0.f);
    atomicAdd(&pooled[t], acc * (1.0f / (float)N));
}

// head: out = gelu(pooled@Wv1+bv1) @ Wv2 + bv2   (single block, 128 threads)
__global__ void mlp_head(const float* __restrict__ pooled, const float* __restrict__ Wv1,
                         const float* __restrict__ bv1, const float* __restrict__ Wv2,
                         const float* __restrict__ bv2, float* __restrict__ out) {
    __shared__ float sd[128];
    int t = threadIdx.x;
    float acc = bv1[t];
    for (int c = 0; c < HID; ++c) acc = fmaf(pooled[c], Wv1[c * 128 + t], acc);
    float g = 0.5f * acc * (1.f + erff(acc * 0.70710678118654752f));   // exact gelu
    sd[t] = g * Wv2[t];
    __syncthreads();
    for (int s2 = 64; s2; s2 >>= 1) { if (t < s2) sd[t] += sd[t + s2]; __syncthreads(); }
    if (t == 0) out[0] = sd[0] + bv2[0];
}

// ---------------- host side ----------------
static inline size_t align256(size_t x) { return (x + 255) & ~(size_t)255; }

extern "C" void kernel_launch(void* const* d_in, const int* in_sizes, int n_in,
                              void* d_out, int out_size, void* d_ws, size_t ws_size,
                              hipStream_t stream) {
    const float* x_f32   = (const float*)d_in[0];
    const int*   ei      = (const int*)d_in[1];
    // d_in[2] = edge_attr (unused by reference: GATConv built without edge_dim)
    const float* W1      = (const float*)d_in[3];
    const float* att_s1  = (const float*)d_in[4];
    const float* att_d1  = (const float*)d_in[5];
    const float* b1      = (const float*)d_in[6];
    const float* W2      = (const float*)d_in[7];
    const float* att_s2  = (const float*)d_in[8];
    const float* att_d2  = (const float*)d_in[9];
    const float* b2      = (const float*)d_in[10];
    const float* Wv1     = (const float*)d_in[11];
    const float* bv1     = (const float*)d_in[12];
    const float* Wv2     = (const float*)d_in[13];
    const float* bv2     = (const float*)d_in[14];
    float* outp          = (float*)d_out;

    const int N    = in_sizes[0] / F_IN;     // 10000 (divisible by 16)
    const int E    = in_sizes[1] / 2;        // 160000
    const int Etot = E + N;                  // self-loops appended

    // ---- carve workspace ----
    char* ws = (char*)d_ws; size_t off = 0;
    auto carve = [&](size_t bytes) { char* p = ws + off; off += align256(bytes); return (void*)p; };
    __hip_bfloat16* Xb   = (__hip_bfloat16*)carve((size_t)N * KP1 * 2);
    __hip_bfloat16* W1t  = (__hip_bfloat16*)carve((size_t)HC1 * KP1 * 2);
    float*    h1   = (float*)carve((size_t)N * HC1 * 4);
    float*    as1  = (float*)carve((size_t)N * HEADS * 4);
    float*    ad1  = (float*)carve((size_t)N * HEADS * 4);
    float*    eb1  = (float*)carve((size_t)Etot * HEADS * 4);
    unsigned* me1  = (unsigned*)carve((size_t)N * HEADS * 4);
    float*    dn1  = (float*)carve((size_t)N * HEADS * 4);
    float*    o1   = (float*)carve((size_t)N * HC1 * 4);
    __hip_bfloat16* X2b  = (__hip_bfloat16*)carve((size_t)N * HC1 * 2);
    __hip_bfloat16* W2t  = (__hip_bfloat16*)carve((size_t)HID * HC1 * 2);
    float*    h2   = (float*)carve((size_t)N * HID * 4);
    float*    as2  = (float*)carve((size_t)N * 4);
    float*    ad2  = (float*)carve((size_t)N * 4);
    float*    eb2  = (float*)carve((size_t)Etot * 4);
    unsigned* me2  = (unsigned*)carve((size_t)N * 4);
    float*    dn2  = (float*)carve((size_t)N * 4);
    float*    o2   = (float*)carve((size_t)N * HID * 4);
    float*    pooled = (float*)carve((size_t)HID * 4);
    (void)ws_size; (void)n_in; (void)out_size;

    auto nb = [](int n, int bs) { return (n + bs - 1) / bs; };

    // ---- zero/init all accumulators (deterministic per call) ----
    fill_f32<<<nb(N * HC1, 256), 256, 0, stream>>>(o1, 0.f, N * HC1);
    fill_f32<<<nb(N * HID, 256), 256, 0, stream>>>(o2, 0.f, N * HID);
    fill_u32<<<nb(N * HEADS, 256), 256, 0, stream>>>(me1, 0u, N * HEADS);
    fill_u32<<<nb(N, 256), 256, 0, stream>>>(me2, 0u, N);
    fill_f32<<<nb(N * HEADS, 256), 256, 0, stream>>>(dn1, 0.f, N * HEADS);
    fill_f32<<<nb(N, 256), 256, 0, stream>>>(dn2, 0.f, N);
    fill_f32<<<1, HID, 0, stream>>>(pooled, 0.f, HID);

    // ---- layer 1: GEMM (bf16 WMMA) ----
    cast_pad_x<<<nb(N * KP1, 256), 256, 0, stream>>>(x_f32, Xb, N);
    transpose_cast_w<<<nb(HC1 * KP1, 256), 256, 0, stream>>>(W1, W1t, F_IN, KP1, HC1);
    {   // waves = (N/16)*(HC1/64) ; 4 waves/block
        int waves = (N / 16) * (HC1 / 64);
        gemm_wmma_bf16<<<nb(waves, 4), 128, 0, stream>>>(
            (const __bf16*)Xb, (const __bf16*)W1t, h1, N, HC1, KP1);
    }
    // ---- layer 1: attention + segment softmax + aggregation ----
    att_dots<<<N, 32 * HEADS, 0, stream>>>(h1, att_s1, att_d1, HEADS, HID, as1, ad1);
    edge_logit_max<<<nb(Etot * HEADS, 256), 256, 0, stream>>>(as1, ad1, ei, E, N, HEADS, eb1, me1);
    edge_exp_sum<<<nb(Etot * HEADS, 256), 256, 0, stream>>>(ei, E, N, HEADS, eb1, me1, dn1);
    aggregate<<<Etot, HC1 / 4, 0, stream>>>(h1, eb1, dn1, ei, E, HEADS, HID, o1);
    bias_relu_cast<<<nb(N * HC1, 256), 256, 0, stream>>>(o1, b1, X2b, N * HC1, HC1);

    // ---- layer 2: GEMM (bf16 WMMA) ----
    transpose_cast_w<<<nb(HID * HC1, 256), 256, 0, stream>>>(W2, W2t, HC1, HC1, HID);
    {
        int waves = (N / 16) * (HID / 64);
        gemm_wmma_bf16<<<nb(waves, 4), 128, 0, stream>>>(
            (const __bf16*)X2b, (const __bf16*)W2t, h2, N, HID, HC1);
    }
    // ---- layer 2: attention + softmax + aggregation ----
    att_dots<<<N, 32, 0, stream>>>(h2, att_s2, att_d2, 1, HID, as2, ad2);
    edge_logit_max<<<nb(Etot, 256), 256, 0, stream>>>(as2, ad2, ei, E, N, 1, eb2, me2);
    edge_exp_sum<<<nb(Etot, 256), 256, 0, stream>>>(ei, E, N, 1, eb2, me2, dn2);
    aggregate<<<Etot, HID / 4, 0, stream>>>(h2, eb2, dn2, ei, E, 1, HID, o2);

    // ---- pool + MLP head ----
    pool_mean_relu<<<256, HID, 0, stream>>>(o2, b2, N, pooled);
    mlp_head<<<1, 128, 0, stream>>>(pooled, Wv1, bv1, Wv2, bv2, outp);
}
